// EP_F_17738214932588
// MI455X (gfx1250) — matrix operand order repaired and checked by reading, not compiled, not run on hardware
//
#include <hip/hip_runtime.h>

typedef __attribute__((ext_vector_type(16))) _Float16 v16h;
typedef __attribute__((ext_vector_type(8)))  _Float16 v8h;
typedef __attribute__((ext_vector_type(2)))  _Float16 h2v;
typedef __attribute__((ext_vector_type(2)))  __fp16   fp16x2;
typedef __attribute__((ext_vector_type(8)))  float    v8f;
typedef int v4i __attribute__((vector_size(4 * sizeof(int))));

namespace cfg {
constexpr int B = 32, A = 96, M = 256, D = 128, H = 8, KD = 16;
constexpr int NMODES = 6, NPRED = 12;
constexpr int NA = B * A;          // 3072 agent tokens
constexpr int NM = B * M;          // 8192 map tokens
constexpr int N2 = NA * NMODES;    // 18432 mode rows
}

#if __has_builtin(__builtin_amdgcn_global_load_async_to_lds_b128)
#define ASYNC_LDS 1
#endif

// 16B global->LDS copy: async-to-LDS on CDNA5 (ASYNCcnt), sync fallback
static __device__ __forceinline__ void ldscpy16(void* dst, const _Float16* src) {
#ifdef ASYNC_LDS
  __builtin_amdgcn_global_load_async_to_lds_b128(
      (__attribute__((address_space(1))) v4i*)const_cast<_Float16*>(src),
      (__attribute__((address_space(3))) v4i*)dst,
      0, 0);
#else
  *(v8h*)dst = *(const v8h*)src;
#endif
}

static __device__ __forceinline__ void wait_async_lds() {
#ifdef ASYNC_LDS
#if __has_builtin(__builtin_amdgcn_s_wait_asynccnt)
  __builtin_amdgcn_s_wait_asynccnt(0);
#else
  asm volatile("s_wait_asynccnt 0x0" ::: "memory");
#endif
#endif
}

// v_cvt_pk_rtz_f16_f32 wrapper (builtin returns __fp16x2; bit-cast to _Float16x2)
static __device__ __forceinline__ h2v pkh(float a, float b) {
  fp16x2 t = __builtin_amdgcn_cvt_pkrtz(a, b);
  return __builtin_bit_cast(h2v, t);
}

// concat two v8h into one v16h (pure register renaming)
static __device__ __forceinline__ v16h cat8(v8h lo, v8h hi) {
  return __builtin_shufflevector(lo, hi, 0, 1, 2, 3, 4, 5, 6, 7,
                                 8, 9, 10, 11, 12, 13, 14, 15);
}

// load 8 consecutive f32 (16B-aligned) -> packed v8h via v_cvt_pk_rtz
static __device__ __forceinline__ v8h cvt8(const float* __restrict__ p) {
  float4 f0 = *(const float4*)p;
  float4 f1 = *(const float4*)(p + 4);
  union { v8h v; h2v h[4]; } u;
  u.h[0] = pkh(f0.x, f0.y);
  u.h[1] = pkh(f0.z, f0.w);
  u.h[2] = pkh(f1.x, f1.y);
  u.h[3] = pkh(f1.z, f1.w);
  return u.v;
}

// weight pre-transpose + f16 convert: Wt[c][k] = W[k][c]
__global__ void k_wt(const float* __restrict__ W, _Float16* __restrict__ Wt,
                     int K, int Nc)
{
  int i = blockIdx.x * blockDim.x + threadIdx.x;
  if (i >= K * Nc) return;
  int c = i / K, k = i % K;
  Wt[i] = (_Float16)W[(size_t)k * Nc + c];
}

// ---------------------------------------------------------------------------
// WMMA GEMM: out[Nr x Nc] = A[Nr x K] @ W[K x Nc] + bias.   K % 32 == 0.
// Weights are pre-transposed f16 [Nc][K], so the B tile is contiguous 16B
// chunks -> staged with async global->LDS copies, double-buffered.
// flags: bit0 = relu, bit1 = elementwise add of `add` buffer.
// ---------------------------------------------------------------------------
__global__ __launch_bounds__(128) void k_gemm(
    const float* __restrict__ Am, const _Float16* __restrict__ WmT,
    const float* __restrict__ bias, const float* __restrict__ add,
    float* __restrict__ out, int Nr, int K, int Nc, int flags)
{
  __shared__ _Float16 As[64][32];       // [row][k]
  __shared__ _Float16 BsT[2][64][32];   // [buf][col][k]
  const int row0 = blockIdx.x * 64, col0 = blockIdx.y * 64;
  const int tid = threadIdx.x, wave = tid >> 5, lane = tid & 31;
  const int lane15 = lane & 15;
  const int a_chunk = (lane < 16) ? 0 : 1;
  const int kb2     = (lane < 16) ? 0 : 2;
  const int arow    = wave * 16 + lane15;

  v8f acc0 = {}, acc1 = {}, acc2 = {}, acc3 = {};

  auto stageB = [&](int kk0, int buf) {
#pragma unroll
    for (int it = 0; it < 2; it++) {
      int ci = tid + it * 128;
      int col = ci >> 2, kg = ci & 3;
      int gc = col0 + col;
      if (gc < Nc) {
        ldscpy16(&BsT[buf][col][kg * 8], WmT + (size_t)gc * K + kk0 + kg * 8);
      } else {
        v8h z = {};
        *(v8h*)&BsT[buf][col][kg * 8] = z;
      }
    }
  };

  stageB(0, 0);
  int cur = 0;
  for (int k0 = 0; k0 < K; k0 += 32) {
    // stage A tile (f32 -> f16 convert, vector loads) + prefetch next tile
#pragma unroll
    for (int it = 0; it < 2; it++) {
      int ci = tid + it * 128;
      int r = ci >> 2, kg = ci & 3;
      int gr = row0 + r;
      v8h val = {};
      if (gr < Nr) {
        const float* ap = Am + (size_t)gr * K + k0 + kg * 8;
        val = cvt8(ap);
        if (k0 + 32 < K) __builtin_prefetch(ap + 32, 0, 3);
      }
      *(v8h*)&As[r][kg * 8] = val;
    }
    wait_async_lds();
    __syncthreads();
    if (k0 + 32 < K) stageB(k0 + 32, cur ^ 1);   // overlap next weight tile

    const v8h* ap = (const v8h*)&As[arow][0];
    v16h af = cat8(ap[kb2], ap[kb2 + 1]);
    const v8h* b0 = (const v8h*)&BsT[cur][0 * 16 + lane15][0];
    acc0 = __builtin_amdgcn_wmma_f32_16x16x32_f16(false, af, false, cat8(b0[kb2], b0[kb2 + 1]), (short)0, acc0, false, false);
    const v8h* b1 = (const v8h*)&BsT[cur][1 * 16 + lane15][0];
    acc1 = __builtin_amdgcn_wmma_f32_16x16x32_f16(false, af, false, cat8(b1[kb2], b1[kb2 + 1]), (short)0, acc1, false, false);
    const v8h* b2 = (const v8h*)&BsT[cur][2 * 16 + lane15][0];
    acc2 = __builtin_amdgcn_wmma_f32_16x16x32_f16(false, af, false, cat8(b2[kb2], b2[kb2 + 1]), (short)0, acc2, false, false);
    const v8h* b3 = (const v8h*)&BsT[cur][3 * 16 + lane15][0];
    acc3 = __builtin_amdgcn_wmma_f32_16x16x32_f16(false, af, false, cat8(b3[kb2], b3[kb2 + 1]), (short)0, acc3, false, false);
    __syncthreads();
    cur ^= 1;
  }

  const int rbase = row0 + wave * 16 + ((lane >= 16) ? 8 : 0);
  v8f* accs[4] = {&acc0, &acc1, &acc2, &acc3};
#pragma unroll
  for (int ct = 0; ct < 4; ct++) {
    int gc = col0 + ct * 16 + lane15;
    if (gc >= Nc) continue;
    float bv = bias ? bias[gc] : 0.f;
#pragma unroll
    for (int i = 0; i < 8; i++) {
      int gr = rbase + i;
      if (gr >= Nr) continue;
      float v = (*accs[ct])[i] + bv;
      if (flags & 1) v = fmaxf(v, 0.f);
      if (flags & 2) v += add[(size_t)gr * Nc + gc];
      out[(size_t)gr * Nc + gc] = v;
    }
  }
}

// ---------------------------------------------------------------------------
// Fused masked-softmax attention, one wave32 per (batch, head, 16-query tile).
// QK^T: K=16 zero-padded into K=32 WMMA.  P@V: full K=32 (32 keys / wmma).
// ---------------------------------------------------------------------------
__global__ __launch_bounds__(32) void k_attention(
    const float* __restrict__ Q, const float* __restrict__ Kb,
    const float* __restrict__ Vb, const unsigned char* __restrict__ mask,
    float* __restrict__ O, int Bn, int T, int S)
{
  __shared__ float     sc[16][256];
  __shared__ _Float16  pb[16][256];
  __shared__ _Float16  VsT[16][32];   // [kd][key_in_tile]
  const int ntq = T >> 4;
  const int bid = blockIdx.x;
  const int qt  = bid % ntq;
  const int h   = (bid / ntq) & 7;
  const int b   = bid / (ntq * 8);
  const int lane = threadIdx.x;
  const int lane15 = lane & 15;
  const int base0 = (lane < 16) ? 0 : 8;
  const int kb2   = (lane < 16) ? 0 : 2;
  const float scale = 0.25f;   // 1/sqrt(KD)

  v16h qf;
  {
    const int qrow = qt * 16 + lane15;
    const float* qp = Q + ((size_t)(b * T + qrow)) * 128 + h * 16 + base0;
    v8h z = {};
    qf = cat8(cvt8(qp), z);
  }

  const int nst = S >> 4;
  for (int st = 0; st < nst; st++) {
    const int key = st * 16 + lane15;
    const float* kp = Kb + ((size_t)(b * S + key)) * 128 + h * 16;
    v8h klo = {}, khi = {};
    klo = cvt8(kp + base0);
    v16h kf = cat8(klo, khi);
    v8f acc = {};
    acc = __builtin_amdgcn_wmma_f32_16x16x32_f16(false, qf, false, kf, (short)0, acc, false, false);
    const int keyg = st * 16 + lane15;
#pragma unroll
    for (int i = 0; i < 8; i++) {
      int r  = i + ((lane >= 16) ? 8 : 0);
      int qg = qt * 16 + r;
      unsigned char m = mask[((size_t)b * T + qg) * S + keyg];
      sc[r][keyg] = m ? acc[i] * scale : -1.0e9f;
    }
  }
  __syncthreads();

  {
    const int srow = lane >> 1;
    const int half = S >> 1;
    const int off = (lane & 1) * half;
    float mx = -3.0e38f;
    for (int j = 0; j < half; j++) mx = fmaxf(mx, sc[srow][off + j]);
    mx = fmaxf(mx, __shfl_xor(mx, 1, 32));
    float sum = 0.f;
    for (int j = 0; j < half; j++) {
      float p = __expf(sc[srow][off + j] - mx);
      sum += p; sc[srow][off + j] = p;
    }
    sum += __shfl_xor(sum, 1, 32);
    float inv = 1.f / sum;
    for (int j = 0; j < half; j++)
      pb[srow][off + j] = (_Float16)(sc[srow][off + j] * inv);
  }
  __syncthreads();

  v8f accO = {};
  const int nst2 = S >> 5;
  const v8h* prow_p = (const v8h*)&pb[lane15][0];
  for (int st2 = 0; st2 < nst2; st2++) {
    {
      const int key = st2 * 32 + lane;
      const float* vsrc = Vb + ((size_t)(b * S + key)) * 128 + h * 16;
      float4 a0 = *(const float4*)(vsrc + 0);
      float4 a1 = *(const float4*)(vsrc + 4);
      float4 a2 = *(const float4*)(vsrc + 8);
      float4 a3 = *(const float4*)(vsrc + 12);
      VsT[0][lane]  = (_Float16)a0.x;  VsT[1][lane]  = (_Float16)a0.y;
      VsT[2][lane]  = (_Float16)a0.z;  VsT[3][lane]  = (_Float16)a0.w;
      VsT[4][lane]  = (_Float16)a1.x;  VsT[5][lane]  = (_Float16)a1.y;
      VsT[6][lane]  = (_Float16)a1.z;  VsT[7][lane]  = (_Float16)a1.w;
      VsT[8][lane]  = (_Float16)a2.x;  VsT[9][lane]  = (_Float16)a2.y;
      VsT[10][lane] = (_Float16)a2.z;  VsT[11][lane] = (_Float16)a2.w;
      VsT[12][lane] = (_Float16)a3.x;  VsT[13][lane] = (_Float16)a3.y;
      VsT[14][lane] = (_Float16)a3.z;  VsT[15][lane] = (_Float16)a3.w;
    }
    __syncthreads();
    const int pc = st2 * 4 + (base0 >> 3);
    v16h pf = cat8(prow_p[pc], prow_p[pc + 2]);
    const v8h* vp = (const v8h*)&VsT[lane15][0];
    v16h vf = cat8(vp[kb2 >> 1], vp[(kb2 >> 1) + 2]);
    accO = __builtin_amdgcn_wmma_f32_16x16x32_f16(false, pf, false, vf, (short)0, accO, false, false);
    __syncthreads();
  }
#pragma unroll
  for (int i = 0; i < 8; i++) {
    int r = i + ((lane >= 16) ? 8 : 0);
    O[((size_t)(b * T + qt * 16 + r)) * 128 + h * 16 + lane15] = accO[i];
  }
}

// ---------------------------------------------------------------------------
__global__ __launch_bounds__(128) void k_layernorm(
    const float* __restrict__ in, float* __restrict__ out,
    const float* __restrict__ g, const float* __restrict__ bparm)
{
  __shared__ float red[128];
  const int t = blockIdx.x, f = threadIdx.x;
  float x = in[(size_t)t * 128 + f];
  red[f] = x; __syncthreads();
  for (int s = 64; s > 0; s >>= 1) { if (f < s) red[f] += red[f + s]; __syncthreads(); }
  float mu = red[0] * (1.f / 128.f); __syncthreads();
  float d = x - mu;
  red[f] = d * d; __syncthreads();
  for (int s = 64; s > 0; s >>= 1) { if (f < s) red[f] += red[f + s]; __syncthreads(); }
  float var = red[0] * (1.f / 128.f);
  out[(size_t)t * 128 + f] = d * rsqrtf(var + 1e-5f) * g[f] + bparm[f];
}

__global__ void k_add(const float* __restrict__ a, const float* __restrict__ b,
                      float* __restrict__ o, int n)
{
  int i = blockIdx.x * blockDim.x + threadIdx.x;
  if (i < n) o[i] = a[i] + b[i];
}

__global__ __launch_bounds__(128) void k_pre_agents(
    const float* __restrict__ input_agents, const float* __restrict__ a_origin,
    const float* __restrict__ a_theta, const float* __restrict__ a_type_table,
    const float* __restrict__ Wt1, const float* __restrict__ bt1,
    const float* __restrict__ Wr1, const float* __restrict__ br1,
    const float* __restrict__ We1, const float* __restrict__ be1,
    float* __restrict__ tcps_out, float* __restrict__ h_time,
    float* __restrict__ h_rel, float* __restrict__ h_vec,
    float* __restrict__ aemb)
{
  const int tok = blockIdx.x, f = threadIdx.x;
  const float* ia = input_agents + (size_t)tok * 19;
  __shared__ float cps[16], avec[14], arel[4], atime[2];
  __shared__ int atype;
  if (f < 16) cps[f] = ia[f];
  if (f == 0) {
    float th = a_theta[tok];
    arel[0] = a_origin[tok * 2]; arel[1] = a_origin[tok * 2 + 1];
    arel[2] = sinf(th); arel[3] = cosf(th);
    atime[0] = ia[16]; atime[1] = ia[17];
    atype = (int)ia[18];
  }
  __syncthreads();
  if (f < 14) avec[f] = cps[f + 2] - cps[f];
  if (f < 16) tcps_out[(size_t)tok * 16 + f] = cps[f] - a_origin[tok * 2 + (f & 1)];
  __syncthreads();
  float ht = bt1[f]; for (int k = 0; k < 2;  k++) ht += atime[k] * Wt1[k * 128 + f];
  float hr = br1[f]; for (int k = 0; k < 4;  k++) hr += arel[k]  * Wr1[k * 128 + f];
  float hv = be1[f]; for (int k = 0; k < 14; k++) hv += avec[k]  * We1[k * 128 + f];
  h_time[(size_t)tok * 128 + f] = fmaxf(ht, 0.f);
  h_rel [(size_t)tok * 128 + f] = fmaxf(hr, 0.f);
  h_vec [(size_t)tok * 128 + f] = fmaxf(hv, 0.f);
  aemb  [(size_t)tok * 128 + f] = a_type_table[atype * 128 + f];
}

__global__ __launch_bounds__(128) void k_pre_maps(
    const float* __restrict__ input_map, const float* __restrict__ m_origin,
    const float* __restrict__ m_theta, const float* __restrict__ m_type_table,
    const float* __restrict__ Wm1, const float* __restrict__ bm1,
    const float* __restrict__ Wr1, const float* __restrict__ br1,
    float* __restrict__ h_map, float* __restrict__ h_rel,
    float* __restrict__ memb)
{
  const int tok = blockIdx.x, f = threadIdx.x;
  const float* im = input_map + (size_t)tok * 23;
  __shared__ float cps[22], mrel[4];
  __shared__ int mtype;
  if (f < 22) cps[f] = im[f] - m_origin[tok * 2 + (f & 1)];
  if (f == 0) {
    float th = m_theta[tok];
    mrel[0] = m_origin[tok * 2]; mrel[1] = m_origin[tok * 2 + 1];
    mrel[2] = sinf(th); mrel[3] = cosf(th);
    mtype = (int)im[22];
  }
  __syncthreads();
  float hm = bm1[f]; for (int k = 0; k < 22; k++) hm += cps[k]  * Wm1[k * 128 + f];
  float hr = br1[f]; for (int k = 0; k < 4;  k++) hr += mrel[k] * Wr1[k * 128 + f];
  h_map[(size_t)tok * 128 + f] = fmaxf(hm, 0.f);
  h_rel[(size_t)tok * 128 + f] = fmaxf(hr, 0.f);
  memb [(size_t)tok * 128 + f] = m_type_table[mtype * 128 + f];
}

__global__ void k_rot_others(const float* __restrict__ ydec,
                             const float* __restrict__ theta,
                             float* __restrict__ out, int na)
{
  int i = blockIdx.x * blockDim.x + threadIdx.x;
  if (i >= na * 12) return;
  int tok = i / 12, j = i % 12;
  float th = theta[tok], c = cosf(th), s = sinf(th);
  float y0 = ydec[(size_t)tok * 24 + j * 2];
  float y1 = ydec[(size_t)tok * 24 + j * 2 + 1];
  out[(size_t)tok * 24 + j * 2]     =  y0 * c + y1 * s;
  out[(size_t)tok * 24 + j * 2 + 1] = -y0 * s + y1 * c;
}

// ---------------------------------------------------------------------------
// Host orchestration
// ---------------------------------------------------------------------------
struct DenseP { const float* b; const float* w; };
struct LNP    { const float* b; const float* g; };
struct MhaP   { const float *bk, *bo, *bq, *bv, *wk, *wo, *wq, *wv; };
struct LayerP { DenseP ff1, ff2; LNP ln1, ln2; MhaP mha; };
struct DenseT { const float* b; const _Float16* wt; int K, Nc; };
struct LayerT { DenseT wq, wk, wv, wo, ff1, ff2; LNP ln1, ln2; };

static void gemm(hipStream_t st, const float* A, const DenseT& W,
                 const float* add, float* out, int Nr, int flags)
{
  dim3 g((Nr + 63) / 64, (W.Nc + 63) / 64), blk(128);
  k_gemm<<<g, blk, 0, st>>>(A, W.wt, W.b, add, out, Nr, W.K, W.Nc, flags);
}

static void attn_layer(hipStream_t st, const LayerT& L, float* xq, const float* kv,
                       int Bn, int T, int S, const unsigned char* mask,
                       float* q, float* k, float* v, float* t, float* h, float* xtmp)
{
  const int NT = Bn * T, NS = Bn * S;
  gemm(st, xq, L.wq, nullptr, q, NT, 0);
  gemm(st, kv, L.wk, nullptr, k, NS, 0);
  gemm(st, kv, L.wv, nullptr, v, NS, 0);
  k_attention<<<Bn * 8 * (T / 16), 32, 0, st>>>(q, k, v, mask, t, Bn, T, S);
  gemm(st, t, L.wo, xq, xtmp, NT, 2);                               // + residual
  k_layernorm<<<NT, 128, 0, st>>>(xtmp, xtmp, L.ln1.g, L.ln1.b);
  gemm(st, xtmp, L.ff1, nullptr, h, NT, 1);                         // relu
  gemm(st, h, L.ff2, xtmp, t, NT, 2);                               // + residual
  k_layernorm<<<NT, 128, 0, st>>>(t, xq, L.ln2.g, L.ln2.b);
}

extern "C" void kernel_launch(void* const* d_in, const int* in_sizes, int n_in,
                              void* d_out, int out_size, void* d_ws, size_t ws_size,
                              hipStream_t stream)
{
  (void)in_sizes; (void)n_in; (void)out_size; (void)ws_size;
  using namespace cfg;

  const float* input_agents = (const float*)d_in[0];
  const float* input_map    = (const float*)d_in[1];
  const float* a_origin     = (const float*)d_in[2];
  const float* a_theta      = (const float*)d_in[3];
  const float* m_origin     = (const float*)d_in[4];
  const float* m_theta      = (const float*)d_in[5];

  // params: jax tree-flatten order (dict keys sorted, lists in sequence)
  int c = 6;
  auto nf      = [&]() { return (const float*)d_in[c++]; };
  auto rdDense = [&]() { DenseP d; d.b = nf(); d.w = nf(); return d; };
  auto rdLN    = [&]() { LNP l; l.b = nf(); l.g = nf(); return l; };
  auto rdMha   = [&]() { MhaP m; m.bk = nf(); m.bo = nf(); m.bq = nf(); m.bv = nf();
                         m.wk = nf(); m.wo = nf(); m.wq = nf(); m.wv = nf(); return m; };
  auto rdLayer = [&]() { LayerP L; L.ff1 = rdDense(); L.ff2 = rdDense();
                         L.ln1 = rdLN(); L.ln2 = rdLN(); L.mha = rdMha(); return L; };

  DenseP a_time_mlp[2];  a_time_mlp[0] = rdDense(); a_time_mlp[1] = rdDense();
  const float* a_type_table = nf();
  DenseP agent_embed[2]; agent_embed[0] = rdDense(); agent_embed[1] = rdDense();
  LayerP agent_map_enc[2][2];
  for (int i = 0; i < 2; i++) for (int l = 0; l < 2; l++) agent_map_enc[i][l] = rdLayer();
  DenseP agent_rel_mlp[2]; agent_rel_mlp[0] = rdDense(); agent_rel_mlp[1] = rdDense();
  LayerP agent_social_enc[2][2];
  for (int i = 0; i < 2; i++) for (int l = 0; l < 2; l++) agent_social_enc[i][l] = rdLayer();
  DenseP dec_mean[3]; for (int i = 0; i < 3; i++) dec_mean[i] = rdDense();
  DenseP dec_oth[3];  for (int i = 0; i < 3; i++) dec_oth[i]  = rdDense();
  LNP layernorm1 = rdLN();
  const float* m_type_table = nf();
  LayerP map_encoder[2]; for (int l = 0; l < 2; l++) map_encoder[l] = rdLayer();
  DenseP mapel_embed[2];   mapel_embed[0]   = rdDense(); mapel_embed[1]   = rdDense();
  DenseP mapel_rel_mlp[2]; mapel_rel_mlp[0] = rdDense(); mapel_rel_mlp[1] = rdDense();
  DenseP multimodal = rdDense();
  DenseP pi[3]; for (int i = 0; i < 3; i++) pi[i] = rdDense();

  const unsigned char* mna  = (const unsigned char*)d_in[c++];  // (B,A,A)
  const unsigned char* mnm  = (const unsigned char*)d_in[c++];  // (B,M,M)
  const unsigned char* mnam = (const unsigned char*)d_in[c++];  // (B,A,M)

  // workspace layout: f32 activations (64 MB) + f16 transposed weights (~4.5 MB)
  float* ws = (float*)d_ws;
  size_t o = 0;
  float* f_aemb = ws + o; o += (size_t)NA * 128;
  float* f_arel = ws + o; o += (size_t)NA * 128;
  float* f_memb = ws + o; o += (size_t)NM * 128;
  float* f_q    = ws + o; o += (size_t)NM * 128;
  float* f_k    = ws + o; o += (size_t)NM * 128;
  float* f_v    = ws + o; o += (size_t)NM * 128;
  float* f_x    = ws + o; o += (size_t)NM * 128;
  float* f_t    = ws + o; o += (size_t)NM * 128;
  float* f_h    = ws + o; o += (size_t)NM * 512;   // FFN hidden / multi [NA x 768]
  float* f_s1   = ws + o; o += (size_t)NA * 128;
  float* f_s2   = ws + o; o += (size_t)NA * 128;
  float* f_d1   = ws + o; o += (size_t)N2 * 128;
  float* f_d2   = ws + o; o += (size_t)N2 * 128;

  _Float16* wtbase = (_Float16*)(ws + o);
  size_t ho = 0;
  auto xw = [&](const float* w, const float* b, int K, int Nc) -> DenseT {
    _Float16* dst = wtbase + ho; ho += (size_t)K * Nc;
    int n = K * Nc;
    k_wt<<<(n + 255) / 256, 256, 0, stream>>>(w, dst, K, Nc);
    DenseT t; t.b = b; t.wt = dst; t.K = K; t.Nc = Nc; return t;
  };
  auto xd = [&](const DenseP& d, int K, int Nc) { return xw(d.w, d.b, K, Nc); };
  auto xlayer = [&](const LayerP& L) {
    LayerT t;
    t.wq  = xw(L.mha.wq, L.mha.bq, 128, 128);
    t.wk  = xw(L.mha.wk, L.mha.bk, 128, 128);
    t.wv  = xw(L.mha.wv, L.mha.bv, 128, 128);
    t.wo  = xw(L.mha.wo, L.mha.bo, 128, 128);
    t.ff1 = xd(L.ff1, 128, 512);
    t.ff2 = xd(L.ff2, 512, 128);
    t.ln1 = L.ln1; t.ln2 = L.ln2;
    return t;
  };

  DenseT t_time2  = xd(a_time_mlp[1], 128, 128);
  DenseT t_aemb2  = xd(agent_embed[1], 128, 128);
  DenseT t_arel2  = xd(agent_rel_mlp[1], 128, 128);
  DenseT t_memb2  = xd(mapel_embed[1], 128, 128);
  DenseT t_mrel2  = xd(mapel_rel_mlp[1], 128, 128);
  LayerT t_map[2];
  for (int l = 0; l < 2; l++) t_map[l] = xlayer(map_encoder[l]);
  LayerT t_am[2][2], t_as[2][2];
  for (int i = 0; i < 2; i++)
    for (int l = 0; l < 2; l++) {
      t_am[i][l] = xlayer(agent_map_enc[i][l]);
      t_as[i][l] = xlayer(agent_social_enc[i][l]);
    }
  DenseT t_do0 = xd(dec_oth[0], 128, 128), t_do1 = xd(dec_oth[1], 128, 128), t_do2 = xd(dec_oth[2], 128, 24);
  DenseT t_dm0 = xd(dec_mean[0], 128, 128), t_dm1 = xd(dec_mean[1], 128, 128), t_dm2 = xd(dec_mean[2], 128, 24);
  DenseT t_pi0 = xd(pi[0], 128, 128), t_pi1 = xd(pi[1], 128, 128), t_pi2 = xd(pi[2], 128, 1);
  DenseT t_mm  = xd(multimodal, 128, 768);

  float* out        = (float*)d_out;
  float* out_logits = out;                                  // (B,A,6)
  float* out_mean   = out + 18432;                          // (B,A,6,24)
  float* out_others = out + 18432 + 442368;                 // (B,A,1,24)
  float* out_tcps   = out + 18432 + 442368 + 73728;         // (B,A,8,2)

  // ---- Phase A: agent embeddings ----
  k_pre_agents<<<NA, 128, 0, stream>>>(
      input_agents, a_origin, a_theta, a_type_table,
      a_time_mlp[0].w, a_time_mlp[0].b, agent_rel_mlp[0].w, agent_rel_mlp[0].b,
      agent_embed[0].w, agent_embed[0].b,
      out_tcps, f_s1, f_s2, f_t, f_aemb);
  gemm(stream, f_s1, t_time2, f_aemb, f_aemb, NA, 2);
  gemm(stream, f_t,  t_aemb2, f_aemb, f_aemb, NA, 2);
  gemm(stream, f_s2, t_arel2, nullptr, f_arel, NA, 0);

  // ---- Phase B: map embeddings ----
  k_pre_maps<<<NM, 128, 0, stream>>>(
      input_map, m_origin, m_theta, m_type_table,
      mapel_embed[0].w, mapel_embed[0].b, mapel_rel_mlp[0].w, mapel_rel_mlp[0].b,
      f_q, f_k, f_memb);
  gemm(stream, f_q, t_memb2, f_memb, f_memb, NM, 2);
  gemm(stream, f_k, t_mrel2, f_memb, f_memb, NM, 2);

  // ---- Phase C: map self-encoder ----
  for (int l = 0; l < 2; l++)
    attn_layer(stream, t_map[l], f_memb, f_memb, B, M, M, mnm,
               f_q, f_k, f_v, f_t, f_h, f_x);

  // ---- Phase D: stacked agent encoders ----
  for (int i = 0; i < 2; i++) {
    k_add<<<(NA * 128 + 255) / 256, 256, 0, stream>>>(f_aemb, f_arel, f_aemb, NA * 128);
    for (int l = 0; l < 2; l++)
      attn_layer(stream, t_am[i][l], f_aemb, f_memb, B, A, M, mnam,
                 f_q, f_k, f_v, f_t, f_h, f_x);
    for (int l = 0; l < 2; l++)
      attn_layer(stream, t_as[i][l], f_aemb, f_aemb, B, A, A, mna,
                 f_q, f_k, f_v, f_t, f_h, f_x);
  }

  // ---- Phase E: heads ----
  k_layernorm<<<NA, 128, 0, stream>>>(f_aemb, f_aemb, layernorm1.g, layernorm1.b);

  gemm(stream, f_aemb, t_do0, nullptr, f_s1, NA, 1);
  gemm(stream, f_s1,   t_do1, nullptr, f_s2, NA, 1);
  gemm(stream, f_s2,   t_do2, nullptr, f_t,  NA, 0);
  k_rot_others<<<(NA * 12 + 255) / 256, 256, 0, stream>>>(f_t, a_theta, out_others, NA);

  gemm(stream, f_aemb, t_mm, nullptr, f_h, NA, 0);   // [NA x 768] == [N2 x 128]

  gemm(stream, f_h,  t_dm0, nullptr, f_d1, N2, 1);
  gemm(stream, f_d1, t_dm1, nullptr, f_d2, N2, 1);
  gemm(stream, f_d2, t_dm2, nullptr, out_mean, N2, 0);

  gemm(stream, f_h,  t_pi0, nullptr, f_d1, N2, 1);
  gemm(stream, f_d1, t_pi1, nullptr, f_d2, N2, 1);
  gemm(stream, f_d2, t_pi2, nullptr, out_logits, N2, 0);
}
